// HierarchicalAtlasStack_85804856640026
// MI455X (gfx1250) — compile-verified, matrix operand order checked by reading, not run on hardware
//
#include <hip/hip_runtime.h>
#include <hip/hip_bf16.h>
#include <math.h>

// ---------------------------------------------------------------------------
// HierarchicalAtlasStack for MI455X (gfx1250, wave32, WMMA)
//
// Dominant cost: q[b,k] += z_b^T Gamma_k z_b == GEMM (2048 x 65536)@(65536 x 256)
//   = 68.7 GFLOP. Gamma (32MB bf16) is L2-resident (192MB L2) -> compute-bound,
//   so all heavy math runs on v_wmma_f32_16x16x32_bf16.
// A rows are vec(z_b outer z_b), synthesized per-fragment in VGPRs (v_pk_mul_bf16)
// from an LDS z tile. Gamma chunks are software-pipelined into double-buffered
// LDS via CDNA5 async global->LDS loads (ASYNCcnt + s_wait_asynccnt).
// Pipeline loop unrolled x2 (static buffer index) with a running source pointer
// (uniform select increment) to minimize per-chunk SALU/VALU overhead.
// ---------------------------------------------------------------------------

typedef __bf16  bf16_t;
typedef __attribute__((ext_vector_type(16))) __bf16 v16bf;
typedef __attribute__((ext_vector_type(8)))  __bf16 v8bf;
typedef __attribute__((ext_vector_type(8)))  float  v8f;
typedef int async_v4i __attribute__((vector_size(16)));

#if defined(__has_builtin)
# if __has_builtin(__builtin_amdgcn_global_load_async_to_lds_b128)
#  define USE_ASYNC_COPY 1
# endif
#endif
#ifndef USE_ASYNC_COPY
# define USE_ASYNC_COPY 0
#endif

#define B_  2048
#define D_  256
#define K_  256
#define NC_ 64
#define F_  512

// ---- workspace layout (bytes) ----------------------------------------------
#define WS_ZB     ((size_t)0)                      // z bf16        [2048][256]  1 MB
#define WS_FEATB  ((size_t)1048576)                // feat bf16     [2048][512]  2 MB
#define WS_WQZB   ((size_t)3145728)                // W_qz bf16     [256][256]   128 KB
#define WS_WQFB   ((size_t)3276800)                // W_qf bf16     [256][512]   256 KB
#define WS_R2     ((size_t)3538944)                // r2 f32        [2048]       8 KB
#define WS_QPART  ((size_t)3547136)                // q partials f32[4][2048][256] 8 MB
#define WS_GAMMAB ((size_t)11935744)               // gamma bf16    [256][256][256] 32 MB

// ---------------------------------------------------------------------------
__global__ __launch_bounds__(256) void cvt_bf16_kernel(const float* __restrict__ src,
                                                       bf16_t* __restrict__ dst, int n) {
  int i = blockIdx.x * 256 + threadIdx.x;
  if (i < n) dst[i] = (bf16_t)src[i];
}

__global__ __launch_bounds__(256) void r2_kernel(const float* __restrict__ z,
                                                 float* __restrict__ r2) {
  __shared__ float red[256];
  int b = blockIdx.x, t = threadIdx.x;
  float v = z[(size_t)b * D_ + t];
  red[t] = v * v;
  __syncthreads();
#pragma unroll
  for (int off = 128; off >= 1; off >>= 1) {
    if (t < off) red[t] += red[t + off];
    __syncthreads();
  }
  if (t == 0) r2[b] = red[0];
}

// ---------------------------------------------------------------------------
static __device__ inline v16bf make_frag(v8bf lo, v8bf hi) {
  v16bf r;
#pragma unroll
  for (int e = 0; e < 8; ++e) { r[e] = lo[e]; r[e + 8] = hi[e]; }
  return r;
}

static __device__ __forceinline__ void async_copy_b128(const bf16_t* gsrc, bf16_t* ldst) {
#if USE_ASYNC_COPY
  __builtin_amdgcn_global_load_async_to_lds_b128(
      (__attribute__((address_space(1))) async_v4i*)(size_t)gsrc,
      (__attribute__((address_space(3))) async_v4i*)ldst,
      0, 0);
#else
  (void)gsrc; (void)ldst;
#endif
}

static __device__ __forceinline__ void wait_async0() {
#if USE_ASYNC_COPY
# if __has_builtin(__builtin_amdgcn_s_wait_asynccnt)
  __builtin_amdgcn_s_wait_asynccnt(0);
# else
  asm volatile("s_wait_asynccnt 0x0" ::: "memory");
# endif
#endif
}

// Main kernel: WG tile 64(M=b) x 128(N=k), 8 waves, wave w -> (mw=w&3, nw=w>>2),
// wave computes rows [b0+mw*16,+16) x cols [k0+nw*64,+64) (4 WMMA accumulators).
// gridDim.z = 4 splits the Gamma i-dimension; split 0 also does linear GEMMs +
// biases. Partials -> qpart[s] (deterministic, no float atomics).
__global__ __launch_bounds__(256) void atlas_main_kernel(
    const bf16_t* __restrict__ zb,      // [2048][256]
    const bf16_t* __restrict__ gammab,  // [256][256][256]  (k, i, j)
    const bf16_t* __restrict__ featb,   // [2048][512]
    const bf16_t* __restrict__ wqzb,    // [256][256]
    const bf16_t* __restrict__ wqfb,    // [256][512]
    const float* __restrict__ b_qz, const float* __restrict__ b_qf,
    float* __restrict__ qpart)          // [4][2048][256]
{
  // Padded rows -> conflict-free 16B ds_loads (z: +4 banks/row, chunks: +20 banks/row)
  __shared__ bf16_t zt[64][264];        // 33 KB z tile
  __shared__ bf16_t bch[2][128][40];    // 2 x 10 KB B chunks (double buffer)
  __shared__ bf16_t fch[64][40];        // 5 KB feature A chunk

  const int b0 = blockIdx.x * 64;
  const int k0 = blockIdx.y * 128;
  const int s  = blockIdx.z;
  const int t  = threadIdx.x;
  const int wave = t >> 5, lane = t & 31;
  const int mw = wave & 3, nw = wave >> 2;
  const int ln = lane & 15;
  const int rl = mw * 16 + ln;          // local A row (M) for this lane
  const int kh = (lane >> 4) << 3;      // K-half base per ISA 16-bit A/B layout

  // stage the 64x256 bf16 z tile (8 x 16B per thread, fixed trip count)
#pragma unroll
  for (int it = 0; it < 8; ++it) {
    int li = it * 256 + t, row = li >> 5, part = li & 31;
    *(uint4*)&zt[row][part << 3] =
        *(const uint4*)&zb[(size_t)(b0 + row) * D_ + (part << 3)];
  }

  v8f acc[4] = {v8f{0}, v8f{0}, v8f{0}, v8f{0}};

  // per-thread staging coordinates: two 16B pieces of the 128x32 chunk
  const int srow = t >> 2;              // 0..63 (second piece: +64)
  const int spart = (t & 3) << 3;       // bf16 element offset 0/8/16/24

  // chunk m (0..511): i = s*64 + (m&63) (inner), jb = (m>>6)*32 (outer)
  // running source pointer, advanced by uniform select:
  //   +256 elems per chunk, or +32 - 63*256 at each 64-chunk (jb) boundary
  const bf16_t* gp = gammab + (size_t)(k0 + srow) * 65536 +
                     (size_t)(s * 64) * 256 + spart;      // chunk 0

#if !USE_ASYNC_COPY
  uint4 r0, r1;                         // register staging (fallback path)
  r0 = *(const uint4*)gp;
  r1 = *(const uint4*)(gp + (size_t)64 * 65536);
#else
  async_copy_b128(gp, &bch[0][srow][spart]);              // start chunk 0
  async_copy_b128(gp + (size_t)64 * 65536, &bch[0][srow + 64][spart]);
#endif
  gp += 256;                            // -> chunk 1

  v8bf z0 = {}, z1 = {};
  for (int nb = 0; nb < 512; nb += 2) {
#pragma unroll
    for (int u = 0; u < 2; ++u) {       // static double-buffer index
      const int n = nb + u;
#if USE_ASYNC_COPY
      wait_async0();                    // my chunk-n pieces landed in LDS
      __syncthreads();                  // everyone's landed; buf[u^1] reads done
      if (n != 511) {                   // start chunk n+1 -> buf[u^1]
        async_copy_b128(gp, &bch[u ^ 1][srow][spart]);
        async_copy_b128(gp + (size_t)64 * 65536, &bch[u ^ 1][srow + 64][spart]);
        gp += (((n + 2) & 63) != 0) ? 256 : (32 - 63 * 256);   // -> chunk n+2
      }
#else
      *(uint4*)&bch[u][srow][spart]      = r0;   // commit chunk n
      *(uint4*)&bch[u][srow + 64][spart] = r1;
      if (n != 511) {                   // fetch chunk n+1 (overlaps compute n)
        r0 = *(const uint4*)gp;
        r1 = *(const uint4*)(gp + (size_t)64 * 65536);
        __builtin_prefetch(gp + 256, 0, 1);
        gp += (((n + 2) & 63) != 0) ? 256 : (32 - 63 * 256);
      }
      __syncthreads();                  // chunk n visible to all waves
#endif

      // ---- compute on chunk n ----
      const int ii = n & 63;
      if (ii == 0) {                    // uniform: new j-block every 64 chunks
        int jb = (n >> 6) << 5;         // (only hit with u==0: boundaries even)
        z0 = *(const v8bf*)&zt[rl][jb + kh];
        z1 = *(const v8bf*)&zt[rl][jb + kh + 16];
      }
      float zi = (float)zt[rl][(s * 64) + ii];

      v16bf bfr[4];                     // preload ALL B fragments: waits overlap
#pragma unroll
      for (int ns = 0; ns < 4; ++ns) {
        int kl = nw * 64 + ns * 16 + ln;
        bfr[ns] = make_frag(*(const v8bf*)&bch[u][kl][kh],
                            *(const v8bf*)&bch[u][kl][kh + 16]);
      }
      v16bf af;                         // A[b, i*256+j] = z[b,i]*z[b,j]
#pragma unroll
      for (int e = 0; e < 8; ++e) {
        af[e]     = (bf16_t)(zi * (float)z0[e]);
        af[e + 8] = (bf16_t)(zi * (float)z1[e]);
      }
#pragma unroll
      for (int ns = 0; ns < 4; ++ns)
        acc[ns] = __builtin_amdgcn_wmma_f32_16x16x32_bf16(
            false, af, false, bfr[ns], (short)0, acc[ns], false, false);
    }
  }

  // ---- linear terms, split 0 only (24 chunks total; simple single-buffer) ----
  if (s == 0) {
    // q += z @ W_qz^T : A = z tile (already in LDS), B[d,k] = W_qz[k,d]
    for (int db = 0; db < 256; db += 32) {
      __syncthreads();
      { int row = t >> 2;
        *(uint4*)&bch[0][row][spart] =
            *(const uint4*)&wqzb[(size_t)(k0 + row) * D_ + db + spart];
        *(uint4*)&bch[0][row + 64][spart] =
            *(const uint4*)&wqzb[(size_t)(k0 + row + 64) * D_ + db + spart]; }
      __syncthreads();
      v16bf af = make_frag(*(const v8bf*)&zt[rl][db + kh],
                           *(const v8bf*)&zt[rl][db + kh + 16]);
      v16bf bfr[4];
#pragma unroll
      for (int ns = 0; ns < 4; ++ns) {
        int kl = nw * 64 + ns * 16 + ln;
        bfr[ns] = make_frag(*(const v8bf*)&bch[0][kl][kh],
                            *(const v8bf*)&bch[0][kl][kh + 16]);
      }
#pragma unroll
      for (int ns = 0; ns < 4; ++ns)
        acc[ns] = __builtin_amdgcn_wmma_f32_16x16x32_bf16(
            false, af, false, bfr[ns], (short)0, acc[ns], false, false);
    }
    // q += features @ W_qf^T
    for (int fb = 0; fb < 512; fb += 32) {
      __syncthreads();
      { int row = t >> 2;
        *(uint4*)&bch[0][row][spart] =
            *(const uint4*)&wqfb[(size_t)(k0 + row) * F_ + fb + spart];
        *(uint4*)&bch[0][row + 64][spart] =
            *(const uint4*)&wqfb[(size_t)(k0 + row + 64) * F_ + fb + spart];
        *(uint4*)&fch[row][spart] =
            *(const uint4*)&featb[(size_t)(b0 + row) * F_ + fb + spart]; }
      __syncthreads();
      v16bf af = make_frag(*(const v8bf*)&fch[rl][kh],
                           *(const v8bf*)&fch[rl][kh + 16]);
      v16bf bfr[4];
#pragma unroll
      for (int ns = 0; ns < 4; ++ns) {
        int kl = nw * 64 + ns * 16 + ln;
        bfr[ns] = make_frag(*(const v8bf*)&bch[0][kl][kh],
                            *(const v8bf*)&bch[0][kl][kh + 16]);
      }
#pragma unroll
      for (int ns = 0; ns < 4; ++ns)
        acc[ns] = __builtin_amdgcn_wmma_f32_16x16x32_bf16(
            false, af, false, bfr[ns], (short)0, acc[ns], false, false);
    }
  }

  // ---- store partials: D layout (M = vgpr + (lane>>4)*8, N = lane&15) ----
#pragma unroll
  for (int ns = 0; ns < 4; ++ns) {
    int col  = k0 + nw * 64 + ns * 16 + ln;
    float bb = (s == 0) ? (b_qz[col] + b_qf[col]) : 0.0f;
#pragma unroll
    for (int v = 0; v < 8; ++v) {
      int row = b0 + mw * 16 + v + ((lane >> 4) << 3);
      qpart[((size_t)s * B_ + row) * K_ + col] = acc[ns][v] + bb;
    }
  }
}

// ---------------------------------------------------------------------------
// Scores / softmax / argmax. One wave per batch row (8 rows per 256-thread WG).
__global__ __launch_bounds__(256) void atlas_scores_kernel(
    const float* __restrict__ z, const float* __restrict__ cq,
    const float* __restrict__ qpart, const float* __restrict__ r2v,
    float* __restrict__ rw_out, int* __restrict__ kc_out)
{
  __shared__ float qrow[8][256];
  __shared__ float zrow[8][256];
  const int b0 = blockIdx.x * 8;
  const int t = threadIdx.x;

#pragma unroll
  for (int it = 0; it < 8; ++it) {
    int li = it * 256 + t, r = li >> 8, c = li & 255;
    size_t idx = (size_t)(b0 + r) * K_ + c;
    qrow[r][c] = qpart[idx] + qpart[(size_t)B_ * K_ + idx] +
                 qpart[2 * (size_t)B_ * K_ + idx] + qpart[3 * (size_t)B_ * K_ + idx];
    zrow[r][c] = z[(size_t)(b0 + r) * D_ + c];
  }
  __syncthreads();

  const int wave = t >> 5, lane = t & 31;
  const int b = b0 + wave;
  const float r2 = r2v[b];
  const float tau = fmaxf(8.0f * fmaxf(1.0f - r2, 1e-3f), 0.01f);  // sqrt(256)*0.5
  const float r2c = fminf(r2, 1.0f - 1e-3f);
  const float lam = 2.0f / (1.0f - r2c + 1e-3f);

  float sc[2];
#pragma unroll
  for (int cc = 0; cc < 2; ++cc) {
    int c = lane * 2 + cc;   // chart id 0..63
    float dzc = 0.f, csq = 0.f, fq = 0.f;
    for (int d = 0; d < 256; ++d) {
      float cv = cq[(size_t)c * K_ + d];  // centers[c][d] == chart_queries[c][d]
      dzc = fmaf(zrow[wave][d], cv, dzc);
      csq = fmaf(cv, cv, csq);
      fq  = fmaf(qrow[wave][d], cv, fq);
    }
    float dist_sq = r2 + csq - 2.0f * dzc;
    float denom   = (1.0f - r2) * (1.0f - csq);
    float arg     = 1.0f + 2.0f * dist_sq / (denom + 1e-3f);
    float dist    = acoshf(fmaxf(arg, 1.0f + 1e-3f));
    sc[cc] = -dist / tau + 0.1f * (fq / lam) / tau;
  }

  // argmax (first-occurrence) + softmax over 64 values (2 per lane)
  float m  = (sc[0] >= sc[1]) ? sc[0] : sc[1];
  int   mi = (sc[0] >= sc[1]) ? lane * 2 : lane * 2 + 1;
#pragma unroll
  for (int off = 16; off >= 1; off >>= 1) {
    float om = __shfl_xor(m, off, 32);
    int   oi = __shfl_xor(mi, off, 32);
    if (om > m || (om == m && oi < mi)) { m = om; mi = oi; }
  }
  float e0 = expf(sc[0] - m), e1 = expf(sc[1] - m);
  float ssum = e0 + e1;
#pragma unroll
  for (int off = 16; off >= 1; off >>= 1) ssum += __shfl_xor(ssum, off, 32);

  rw_out[(size_t)b * NC_ + lane * 2]     = e0 / ssum;
  rw_out[(size_t)b * NC_ + lane * 2 + 1] = e1 / ssum;
  if (lane == 0) kc_out[b] = mi;
}

// ---------------------------------------------------------------------------
extern "C" void kernel_launch(void* const* d_in, const int* in_sizes, int n_in,
                              void* d_out, int out_size, void* d_ws, size_t ws_size,
                              hipStream_t stream) {
  const float* z    = (const float*)d_in[0];   // [2048][256]
  const float* feat = (const float*)d_in[1];   // [2048][512]
  const float* wqz  = (const float*)d_in[2];   // [256][256]
  const float* bqz  = (const float*)d_in[3];   // [256]
  const float* wqf  = (const float*)d_in[4];   // [256][512]
  const float* bqf  = (const float*)d_in[5];   // [256]
  const float* qg   = (const float*)d_in[6];   // [256][256][256]
  const float* cq   = (const float*)d_in[7];   // [64][256]

  char* ws = (char*)d_ws;
  bf16_t* zbp    = (bf16_t*)(ws + WS_ZB);
  bf16_t* featb  = (bf16_t*)(ws + WS_FEATB);
  bf16_t* wqzb   = (bf16_t*)(ws + WS_WQZB);
  bf16_t* wqfb   = (bf16_t*)(ws + WS_WQFB);
  float*  r2     = (float*)(ws + WS_R2);
  float*  qpart  = (float*)(ws + WS_QPART);
  bf16_t* gammab = (bf16_t*)(ws + WS_GAMMAB);

  // Phase 1: precision conversion + r2
  cvt_bf16_kernel<<<(B_ * D_) / 256, 256, 0, stream>>>(z, zbp, B_ * D_);
  cvt_bf16_kernel<<<(B_ * F_) / 256, 256, 0, stream>>>(feat, featb, B_ * F_);
  cvt_bf16_kernel<<<(K_ * D_) / 256, 256, 0, stream>>>(wqz, wqzb, K_ * D_);
  cvt_bf16_kernel<<<(K_ * F_) / 256, 256, 0, stream>>>(wqf, wqfb, K_ * F_);
  cvt_bf16_kernel<<<(K_ * D_ * D_) / 256, 256, 0, stream>>>(qg, gammab, K_ * D_ * D_);
  r2_kernel<<<B_, 256, 0, stream>>>(z, r2);

  // Phase 2: WMMA GEMMs (quadratic form + linear terms) -> qpart[4]
  dim3 grid(B_ / 64, K_ / 128, 4);
  atlas_main_kernel<<<grid, 256, 0, stream>>>(zbp, gammab, featb, wqzb, wqfb,
                                              bqz, bqf, qpart);

  // Phase 3: hyperbolic distances, scores, softmax, argmax
  float* rw_out = (float*)d_out;                    // [2048][64]
  int*   kc_out = (int*)((float*)d_out + B_ * NC_); // [2048] int32 bits
  atlas_scores_kernel<<<B_ / 8, 256, 0, stream>>>(z, cq, qpart, r2, rw_out, kc_out);
}
